// MultiheadFlashDiff2_14791867367882
// MI455X (gfx1250) — matrix operand order repaired and checked by reading, not compiled, not run on hardware
//
#include <hip/hip_runtime.h>
#include <hip/hip_bf16.h>
#include <math.h>

typedef __bf16 bf16;
typedef bf16  v16bf __attribute__((ext_vector_type(16)));
typedef bf16  v8bf  __attribute__((ext_vector_type(8)));
typedef bf16  bf16x4 __attribute__((ext_vector_type(4)));
typedef float v8f   __attribute__((ext_vector_type(8)));
typedef float f32x4 __attribute__((ext_vector_type(4)));

#define T_SEQ 2048
#define E_DIM 2048
#define H_HEADS 16
#define D_HEAD 64
#define LAMBDA_INIT 0.783605766531f
#define NEG_BIG -1e30f

// ---------------------------------------------------------------------------
// WMMA fragment loaders (this file's contiguous k order, wave32)
//  A (16x32 bf16): lane l -> M = l%16 ; element e -> K = (l/16)*16 + e
//  B (32x16 bf16) from a TRANSPOSED tile Bt[n][k]: lane l -> N = l%16 ;
//    element e -> K = (l/16)*16 + e  => one contiguous 16-element (32B) LDS load.
// ---------------------------------------------------------------------------
__device__ inline v16bf load_a_frag(const bf16* base, int pitch, int row, int k0, int lane) {
  const bf16* p = base + row * pitch + k0 + ((lane >> 4) & 1) * 16;
  v8bf lo = *(const v8bf*)(p);
  v8bf hi = *(const v8bf*)(p + 8);
  v16bf a;
#pragma unroll
  for (int i = 0; i < 8; ++i) { a[i] = lo[i]; a[i + 8] = hi[i]; }
  return a;
}

__device__ inline v16bf load_b_frag(const bf16* base, int pitch, int nrow, int k0, int lane) {
  const bf16* p = base + nrow * pitch + k0 + ((lane >> 4) & 1) * 16;
  return *(const v16bf*)(p);
}

__device__ inline v8f wmma_bf16(v16bf a, v16bf b, v8f c) {
  return __builtin_amdgcn_wmma_f32_16x16x32_bf16(false, a, false, b, (short)0, c, false, false);
}

// Async global->LDS copy (CDNA5, tracked by ASYNCcnt; bypasses VGPRs)
__device__ inline void async_lds_b128(unsigned int lds_byte, const void* gaddr) {
  asm volatile("global_load_async_to_lds_b128 %0, %1, off"
               :: "v"(lds_byte), "v"((unsigned long long)(uintptr_t)gaddr)
               : "memory");
}
__device__ inline void wait_asynccnt0() {
  asm volatile("s_wait_asynccnt 0" ::: "memory");
}
__device__ inline unsigned int lds_addr32(const void* p) {
  return (unsigned int)(uintptr_t)p;   // low 32 bits of shared aperture = LDS offset
}

// ---------------------------------------------------------------------------
// Generic row-major GEMM: C(MxN,f32) = A(MxK,f32) x B(KxN,f32), bf16 WMMA math.
// Block 256 thr (8 waves), tile 128x128x32; each wave computes 32x64.
// LDS double-buffered; tile staging uses batched b128 global loads so a single
// loadcnt wait covers the whole batch (overlaps with current tile's WMMAs).
// ---------------------------------------------------------------------------
#define GP 48  // LDS pitch (96B rows, 32B aligned)

__global__ __launch_bounds__(256)
void gemm_bf16wmma(const float* __restrict__ A, const float* __restrict__ B,
                   float* __restrict__ C, int M, int N, int K) {
  __shared__ __attribute__((aligned(32))) bf16 As[2][128 * GP];
  __shared__ __attribute__((aligned(32))) bf16 Bt[2][128 * GP];
  const int tid  = threadIdx.x;
  const int lane = tid & 31;
  const int wave = tid >> 5;
  const int wm   = (wave >> 1) * 32;
  const int wn   = (wave & 1) * 64;
  const int m0   = blockIdx.y * 128;
  const int n0   = blockIdx.x * 128;

  v8f acc[2][4];
#pragma unroll
  for (int i = 0; i < 2; ++i)
#pragma unroll
    for (int j = 0; j < 4; ++j) acc[i][j] = (v8f)0.0f;

  auto load_tile = [&](int k0, int buf) {
    f32x4 av[4], bv[4];
    // A: 128x32 floats = 1024 float4 chunks; 4 per thread (8 chunks per row)
#pragma unroll
    for (int i = 0; i < 4; ++i) {
      int c4 = i * 256 + tid;
      int r = c4 >> 3, c = (c4 & 7) << 2;
      av[i] = *(const f32x4*)&A[(m0 + r) * K + k0 + c];
    }
    // B: 32x128 floats = 1024 float4 chunks; 4 per thread (32 chunks per row)
#pragma unroll
    for (int i = 0; i < 4; ++i) {
      int c4 = i * 256 + tid;
      int r = c4 >> 5, c = (c4 & 31) << 2;
      bv[i] = *(const f32x4*)&B[(k0 + r) * N + n0 + c];
    }
    // convert + store (A: packed b64 store; B: transposed scalar stores)
#pragma unroll
    for (int i = 0; i < 4; ++i) {
      int c4 = i * 256 + tid;
      int r = c4 >> 3, c = (c4 & 7) << 2;
      bf16x4 hx;
#pragma unroll
      for (int j = 0; j < 4; ++j) hx[j] = (bf16)av[i][j];
      *(bf16x4*)&As[buf][r * GP + c] = hx;
    }
#pragma unroll
    for (int i = 0; i < 4; ++i) {
      int c4 = i * 256 + tid;
      int r = c4 >> 5, c = (c4 & 31) << 2;
#pragma unroll
      for (int j = 0; j < 4; ++j) Bt[buf][(c + j) * GP + r] = (bf16)bv[i][j];
    }
  };

  const int nsteps = K >> 5;
  load_tile(0, 0);
  __syncthreads();

  for (int s = 0; s < nsteps; ++s) {
    const int cur = s & 1;
    if (s + 1 < nsteps) {
      __builtin_prefetch(&A[(m0 + (tid >> 1)) * K + ((s + 2) << 5)], 0, 1);
      load_tile((s + 1) << 5, cur ^ 1);      // fill other buffer while computing
    }
    v16bf afrag[2], bfrag[4];
#pragma unroll
    for (int ms = 0; ms < 2; ++ms)
      afrag[ms] = load_a_frag(As[cur], GP, wm + ms * 16 + (lane & 15), 0, lane);
#pragma unroll
    for (int ns = 0; ns < 4; ++ns)
      bfrag[ns] = load_b_frag(Bt[cur], GP, wn + ns * 16 + (lane & 15), 0, lane);
#pragma unroll
    for (int ms = 0; ms < 2; ++ms)
#pragma unroll
      for (int ns = 0; ns < 4; ++ns)
        acc[ms][ns] = wmma_bf16(afrag[ms], bfrag[ns], acc[ms][ns]);
    __syncthreads();
  }

#pragma unroll
  for (int ms = 0; ms < 2; ++ms)
#pragma unroll
    for (int ns = 0; ns < 4; ++ns)
#pragma unroll
      for (int r = 0; r < 8; ++r) {
        int row = m0 + wm + ms * 16 + r + ((lane >> 4) << 3);
        int col = n0 + wn + ns * 16 + (lane & 15);
        C[row * N + col] = acc[ms][ns][r];
      }
}

// ---------------------------------------------------------------------------
// Rotary (interleaved) fp32 -> bf16.  Layout (T, 2H, D), D=64, 32 freq pairs.
// ---------------------------------------------------------------------------
__global__ __launch_bounds__(256)
void rotary_to_bf16(const float* __restrict__ src, bf16* __restrict__ dst) {
  int idx = blockIdx.x * blockDim.x + threadIdx.x;   // T * 32 * 32 threads
  int j  = idx & 31;
  int ch = (idx >> 5) & 31;
  int t  = idx >> 10;
  float ang = (float)t * __expf(-(float)j * (9.21034037198f / 32.f)); // 10000^(-j/32)
  float s, c;
  __sincosf(ang, &s, &c);
  const float* p = src + t * E_DIM + ch * 64 + 2 * j;
  float x1 = p[0], x2 = p[1];
  dst[t * E_DIM + ch * 64 + 2 * j]     = (bf16)(x1 * c - x2 * s);
  dst[t * E_DIM + ch * 64 + 2 * j + 1] = (bf16)(x1 * s + x2 * c);
}

// ---------------------------------------------------------------------------
// Flash differential attention. grid (T/64, H, pair), block 128 (4 waves).
// Q/K tiles are raw bf16 copies -> async global->LDS DMA (ASYNCcnt).
// V tile: batched float4 loads, fp32->bf16, transposed store.
// ---------------------------------------------------------------------------
#define AP 80  // LDS pitch (160B rows, 16B-aligned chunks)

__global__ __launch_bounds__(128)
void diff_attn(const bf16* __restrict__ qrot, const bf16* __restrict__ krot,
               const float* __restrict__ vsrc, float* __restrict__ out1,
               float* __restrict__ out2) {
  __shared__ __attribute__((aligned(32))) bf16 Qs[64 * AP];
  __shared__ __attribute__((aligned(32))) bf16 Ks[64 * AP];
  __shared__ __attribute__((aligned(32))) bf16 Vt[128 * AP];  // Vt[d][key]
  __shared__ __attribute__((aligned(32))) bf16 Ps[64 * AP];   // per-wave P scratch

  const int q0  = blockIdx.x * 64;
  const int h   = blockIdx.y;
  const int pr  = blockIdx.z;
  float* out    = pr ? out2 : out1;
  const int qkc = (2 * h + pr) * 64;   // q1/q2,k1/k2 channel base
  const int vc  = h * 128;             // [v1|v2] channel base
  const int tid = threadIdx.x, lane = tid & 31, wave = tid >> 5;
  const float scale = 0.125f;          // D^-0.5

  const unsigned int QsB = lds_addr32(Qs);
  const unsigned int KsB = lds_addr32(Ks);

  // Q tile 64x64 bf16: 512 x 16B chunks via async DMA
#pragma unroll
  for (int i = 0; i < 4; ++i) {
    int idx = i * 128 + tid;
    int r = idx >> 3, c8 = idx & 7;
    async_lds_b128(QsB + r * (AP * 2) + c8 * 16,
                   qrot + (q0 + r) * E_DIM + qkc + c8 * 8);
  }

  v8f oacc[8];
#pragma unroll
  for (int n = 0; n < 8; ++n) oacc[n] = (v8f)0.0f;
  float mrun[8], lrun[8];
#pragma unroll
  for (int r = 0; r < 8; ++r) { mrun[r] = NEG_BIG; lrun[r] = 0.f; }

  for (int s0 = 0; s0 <= q0; s0 += 64) {
    __syncthreads();                   // all waves done with prev Ks/Vt
#pragma unroll
    for (int i = 0; i < 4; ++i) {      // K tile 64x64 bf16 via async DMA
      int idx = i * 128 + tid;
      int r = idx >> 3, c8 = idx & 7;
      async_lds_b128(KsB + r * (AP * 2) + c8 * 16,
                     krot + (s0 + r) * E_DIM + qkc + c8 * 8);
    }
    // V tile 64x128 fp32: 2048 float4 chunks (32 per row); batched loads,
    // then fp32->bf16 transposed stores.  4 batches of 4 to bound registers.
#pragma unroll
    for (int b = 0; b < 4; ++b) {
      f32x4 tv[4];
#pragma unroll
      for (int i = 0; i < 4; ++i) {
        int idx = (b * 4 + i) * 128 + tid;
        int r = idx >> 5, c = (idx & 31) << 2;
        tv[i] = *(const f32x4*)&vsrc[(s0 + r) * E_DIM + vc + c];
      }
#pragma unroll
      for (int i = 0; i < 4; ++i) {
        int idx = (b * 4 + i) * 128 + tid;
        int r = idx >> 5, c = (idx & 31) << 2;
#pragma unroll
        for (int j = 0; j < 4; ++j) Vt[(c + j) * AP + r] = (bf16)tv[i][j];
      }
    }
    wait_asynccnt0();                  // async K (and first-pass Q) in LDS
    __syncthreads();

    // S = Q K^T  (16x64 per wave; K-dim 64 = 2 wmma steps)
    v8f sacc[4];
#pragma unroll
    for (int ns = 0; ns < 4; ++ns) sacc[ns] = (v8f)0.0f;
    v16bf qf[2];
#pragma unroll
    for (int c = 0; c < 2; ++c)
      qf[c] = load_a_frag(Qs, AP, wave * 16 + (lane & 15), c * 32, lane);
#pragma unroll
    for (int ns = 0; ns < 4; ++ns)
#pragma unroll
      for (int c = 0; c < 2; ++c) {
        v16bf kf = load_b_frag(Ks, AP, ns * 16 + (lane & 15), c * 32, lane);
        sacc[ns] = wmma_bf16(qf[c], kf, sacc[ns]);
      }

    // online softmax per row (C layout: row = r + 8*(lane/16), col = lane%16)
#pragma unroll
    for (int r = 0; r < 8; ++r) {
      int qrow = q0 + wave * 16 + r + ((lane >> 4) << 3);
      float vals[4];
      float mt = NEG_BIG;
#pragma unroll
      for (int ns = 0; ns < 4; ++ns) {
        int key = s0 + ns * 16 + (lane & 15);
        float v = sacc[ns][r] * scale;
        if (key > qrow) v = NEG_BIG;
        vals[ns] = v;
        mt = fmaxf(mt, v);
      }
#pragma unroll
      for (int m = 1; m < 16; m <<= 1) mt = fmaxf(mt, __shfl_xor(mt, m, 32));
      float mnew = fmaxf(mrun[r], mt);
      float corr = __expf(mrun[r] - mnew);
      float ts = 0.f;
#pragma unroll
      for (int ns = 0; ns < 4; ++ns) {
        float p = __expf(vals[ns] - mnew);
        ts += p;
        Ps[(wave * 16 + r + ((lane >> 4) << 3)) * AP + ns * 16 + (lane & 15)] = (bf16)p;
      }
#pragma unroll
      for (int m = 1; m < 16; m <<= 1) ts += __shfl_xor(ts, m, 32);
      lrun[r] = lrun[r] * corr + ts;
      mrun[r] = mnew;
#pragma unroll
      for (int n = 0; n < 8; ++n) oacc[n][r] *= corr;
    }

    // P stores and loads are wave-private: wait DS pipe, no block barrier needed
    asm volatile("s_wait_dscnt 0" ::: "memory");

    // O += P * V   (key dim 64 = 2 wmma steps, 8 output N subtiles)
#pragma unroll
    for (int c = 0; c < 2; ++c) {
      v16bf pf = load_a_frag(Ps + wave * 16 * AP, AP, (lane & 15), c * 32, lane);
#pragma unroll
      for (int n = 0; n < 8; ++n) {
        v16bf vf = load_b_frag(Vt, AP, n * 16 + (lane & 15), c * 32, lane);
        oacc[n] = wmma_bf16(pf, vf, oacc[n]);
      }
    }
  }

#pragma unroll
  for (int r = 0; r < 8; ++r) lrun[r] = 1.f / lrun[r];
#pragma unroll
  for (int n = 0; n < 8; ++n)
#pragma unroll
    for (int r = 0; r < 8; ++r) {
      int row = q0 + wave * 16 + r + ((lane >> 4) << 3);
      int col = vc + n * 16 + (lane & 15);
      out[row * E_DIM + col] = oacc[n][r] * lrun[r];
    }
}

// ---------------------------------------------------------------------------
// Small helper kernels
// ---------------------------------------------------------------------------
__global__ void lambda_kernel(const float* lq1, const float* lk1,
                              const float* lq2, const float* lk2, float* lam) {
  __shared__ float s1[64], s2[64];
  int t = threadIdx.x;
  s1[t] = lq1[t] * lk1[t];
  s2[t] = lq2[t] * lk2[t];
  __syncthreads();
  for (int o = 32; o > 0; o >>= 1) {
    if (t < o) { s1[t] += s1[t + o]; s2[t] += s2[t + o]; }
    __syncthreads();
  }
  if (t == 0) *lam = __expf(s1[0]) - __expf(s2[0]) + LAMBDA_INIT;
}

// h = irfft(gaussian lowpass), length T
__global__ void timekernel(float* hbuf) {
  int n = blockIdx.x * blockDim.x + threadIdx.x;
  if (n >= T_SEQ) return;
  const float sigma = 0.15f;
  float acc = 0.f;
  for (int j = 0; j <= 1024; ++j) {
    float x = (float)j / 1024.f;
    float w = __expf(-0.5f * (x / sigma) * (x / sigma));
    float coef = (j == 0 || j == 1024) ? 1.f : 2.f;
    acc += coef * w * __cosf(6.283185307179586f * (float)j * (float)n / 2048.f);
  }
  hbuf[n] = acc / 2048.f;
}

__global__ void combine_kernel(const float* __restrict__ a1, const float* __restrict__ a2,
                               const float* __restrict__ lam, float* __restrict__ g) {
  int i = blockIdx.x * blockDim.x + threadIdx.x;
  g[i] = a1[i] - (*lam) * a2[i];
}

__global__ void circulant_kernel(const float* __restrict__ hbuf, float* __restrict__ Hm) {
  int i = blockIdx.x * blockDim.x + threadIdx.x;   // T*T
  int t = i >> 11, s = i & 2047;
  Hm[i] = hbuf[(t - s) & 2047];
}

__global__ __launch_bounds__(128)
void rms_kernel(const float* __restrict__ x, const float* __restrict__ w,
                float* __restrict__ y) {
  __shared__ float red[128];
  int row = blockIdx.x, t = threadIdx.x;
  float v = x[row * 128 + t];
  red[t] = v * v;
  __syncthreads();
  for (int o = 64; o > 0; o >>= 1) {
    if (t < o) red[t] += red[t + o];
    __syncthreads();
  }
  float r = rsqrtf(red[0] / 128.f + 1e-5f);
  y[row * 128 + t] = v * r * w[t] * (1.f - LAMBDA_INIT);
}

// ---------------------------------------------------------------------------
extern "C" void kernel_launch(void* const* d_in, const int* in_sizes, int n_in,
                              void* d_out, int out_size, void* d_ws, size_t ws_size,
                              hipStream_t stream) {
  const float* query = (const float*)d_in[0];
  const float* Wq    = (const float*)d_in[1];
  const float* Wk    = (const float*)d_in[2];
  const float* Wv    = (const float*)d_in[3];
  const float* Wout  = (const float*)d_in[4];
  const float* lq1   = (const float*)d_in[5];
  const float* lk1   = (const float*)d_in[6];
  const float* lq2   = (const float*)d_in[7];
  const float* lk2   = (const float*)d_in[8];
  const float* rmsw  = (const float*)d_in[9];

  char* ws = (char*)d_ws;
  float* Qf    = (float*)(ws);                         // 16MB
  float* Kf    = (float*)(ws + (16u << 20));           // 16MB
  float* Vf    = (float*)(ws + (32u << 20));           // 16MB
  bf16*  qrot  = (bf16*) (ws + (48u << 20));           // 8MB
  bf16*  krot  = (bf16*) (ws + (56u << 20));           // 8MB
  float* attn1 = Qf;                                   // reuse after rotary
  float* attn2 = Kf;
  float* g     = (float*)(ws + (48u << 20));           // reuse qrot/krot after attn
  float* Hm    = Vf;                                   // reuse V after attn
  float* convo = attn1;                                // reuse after combine
  float* ynorm = attn2;
  float* lam   = (float*)(ws + (64u << 20));
  float* hbuf  = (float*)(ws + (64u << 20) + 256);

  dim3 gblk(16, 16);
  // 1-3: projections Q,K,V  (bf16 WMMA GEMMs)
  gemm_bf16wmma<<<gblk, 256, 0, stream>>>(query, Wq, Qf, T_SEQ, E_DIM, E_DIM);
  gemm_bf16wmma<<<gblk, 256, 0, stream>>>(query, Wk, Kf, T_SEQ, E_DIM, E_DIM);
  gemm_bf16wmma<<<gblk, 256, 0, stream>>>(query, Wv, Vf, T_SEQ, E_DIM, E_DIM);
  // 4: rotary -> bf16 operands
  rotary_to_bf16<<<8192, 256, 0, stream>>>(Qf, qrot);
  rotary_to_bf16<<<8192, 256, 0, stream>>>(Kf, krot);
  // 5-6: lambda scalar and time-domain filter kernel
  lambda_kernel<<<1, 64, 0, stream>>>(lq1, lk1, lq2, lk2, lam);
  timekernel<<<8, 256, 0, stream>>>(hbuf);
  // 7: two causal attentions (pair in blockIdx.z)
  diff_attn<<<dim3(T_SEQ / 64, H_HEADS, 2), 128, 0, stream>>>(qrot, krot, Vf, attn1, attn2);
  // 8: g = attn1 - lambda*attn2
  combine_kernel<<<(T_SEQ * E_DIM) / 256, 256, 0, stream>>>(attn1, attn2, lam, g);
  // 9-10: spectral filter as circulant GEMM
  circulant_kernel<<<(T_SEQ * T_SEQ) / 256, 256, 0, stream>>>(hbuf, Hm);
  gemm_bf16wmma<<<gblk, 256, 0, stream>>>(Hm, g, convo, T_SEQ, E_DIM, T_SEQ);
  // 11: RMS norm over last dim (128) + (1-lambda_init) scale
  rms_kernel<<<T_SEQ * H_HEADS, 128, 0, stream>>>(convo, rmsw, ynorm);
  // 12: output projection
  gemm_bf16wmma<<<gblk, 256, 0, stream>>>(ynorm, Wout, (float*)d_out, T_SEQ, E_DIM, E_DIM);
}
